// MockGNN_3994319585791
// MI455X (gfx1250) — matrix operand order repaired and checked by reading, not compiled, not run on hardware
//
#include <hip/hip_runtime.h>
#include <hip/hip_bf16.h>

typedef __attribute__((ext_vector_type(16))) _Float16 v16h;
typedef __attribute__((ext_vector_type(8)))  float    v8f;

#define D_FEAT 64

// ---------------------------------------------------------------------------
// 1) deg[i] = 1.0 (self-loop contribution to degree)
__global__ void gcn_deg_init(float* __restrict__ deg, int n) {
    int i = blockIdx.x * blockDim.x + threadIdx.x;
    if (i < n) deg[i] = 1.0f;
}

// 2) deg[dst[e]] += 1 for every edge
__global__ void gcn_deg_count(const long long* __restrict__ ei, float* __restrict__ deg, int E) {
    int e = blockIdx.x * blockDim.x + threadIdx.x;
    if (e < E) {
        int d = (int)ei[(long long)E + e];   // row 1 of edge_index = dst
        unsafeAtomicAdd(&deg[d], 1.0f);
    }
}

// 3) dinv[i] = rsqrt(deg[i])   (deg >= 1 always because of self-loops)
__global__ void gcn_dinv(const float* __restrict__ deg, float* __restrict__ dinv, int n) {
    int i = blockIdx.x * blockDim.x + threadIdx.x;
    if (i < n) dinv[i] = __frsqrt_rn(deg[i]);
}

// ---------------------------------------------------------------------------
// 4) h = x @ W via v_wmma_f32_16x16x32_f16.
//    One wave -> one 16x16 output tile (16 rows of x, 16 output features).
//    Block = 128 threads = 4 waves -> covers all 64 output features of a
//    16-row strip. Grid = n/16 strips. n is a multiple of 16 (65536).
__global__ __launch_bounds__(128) void gcn_gemm_wmma(const float* __restrict__ x,
                                                     const float* __restrict__ W,
                                                     float* __restrict__ h) {
    const int lane = threadIdx.x & 31;
    const int wave = threadIdx.x >> 5;     // 0..3 -> output-feature tile
    const int row0 = blockIdx.x * 16;
    const int n0   = wave * 16;
    const int g    = lane >> 4;            // lane group (0: lanes 0-15, 1: 16-31)
    const int m    = lane & 15;            // row (A) or column (B, C/D) index

    v8f c = {};
    const float* xrow = x + (long long)(row0 + m) * D_FEAT;

    #pragma unroll
    for (int kb = 0; kb < D_FEAT; kb += 32) {
        // A fragment (16x32 f16, ISA layout): lane group g, VGPR v holds
        // K = 2v + 8g (v<4) or 16 + 2(v-4) + 8g (v>=4), two halves each.
        v16h a;
        #pragma unroll
        for (int v = 0; v < 8; ++v) {
            int k = (v < 4) ? (2 * v + 8 * g) : (16 + 2 * (v - 4) + 8 * g);
            a[2 * v]     = (_Float16)xrow[kb + k];
            a[2 * v + 1] = (_Float16)xrow[kb + k + 1];
        }
        // B fragment (32x16 f16): lane group g holds K = 8g + e (e<8) or 16 + 8g + (e-8) (e>=8), column = m.
        v16h bfrag;
        #pragma unroll
        for (int e = 0; e < 16; ++e) {
            int kk = (e < 8) ? (8 * g + e) : (16 + 8 * g + (e - 8));
            bfrag[e] = (_Float16)W[(long long)(kb + kk) * D_FEAT + n0 + m];
        }
        c = __builtin_amdgcn_wmma_f32_16x16x32_f16(
                /*neg_a=*/false, a, /*neg_b=*/false, bfrag,
                /*c_mod=*/(short)0, c, /*reuse_a=*/false, /*reuse_b=*/false);
    }

    // C/D layout: VGPR r -> row (r + 8g), column m.
    #pragma unroll
    for (int r = 0; r < 8; ++r) {
        h[(long long)(row0 + r + 8 * g) * D_FEAT + n0 + m] = c[r];
    }
}

// ---------------------------------------------------------------------------
// 5) out = b + dinv[i]^2 * h   (self-loop message; also initializes out)
__global__ void gcn_selfloop_bias(const float* __restrict__ h,
                                  const float* __restrict__ dinv,
                                  const float* __restrict__ bias,
                                  float* __restrict__ out, int n) {
    long long t = (long long)blockIdx.x * blockDim.x + threadIdx.x;
    if (t < (long long)n * D_FEAT) {
        int i = (int)(t >> 6);
        int f = (int)(t & 63);
        float di = dinv[i];
        out[t] = bias[f] + di * di * h[t];
    }
}

// 6) edge scatter: 16 lanes per edge, each lane owns a float4 feature chunk.
//    Gathers h[src] (L2-resident: 16 MB << 192 MB L2), scales by
//    dinv[src]*dinv[dst], hardware-atomic f32 adds into out[dst].
__global__ void gcn_scatter(const long long* __restrict__ ei,
                            const float* __restrict__ dinv,
                            const float* __restrict__ h,
                            float* __restrict__ out, int E) {
    long long t = (long long)blockIdx.x * blockDim.x + threadIdx.x;
    int e = (int)(t >> 4);
    int c = (int)(t & 15);
    if (e < E) {
        int s = (int)ei[e];
        int d = (int)ei[(long long)E + e];
        float norm = dinv[s] * dinv[d];
        const float4 hv = *(const float4*)(h + (long long)s * D_FEAT + c * 4);
        float* o = out + (long long)d * D_FEAT + c * 4;
        unsafeAtomicAdd(o + 0, norm * hv.x);
        unsafeAtomicAdd(o + 1, norm * hv.y);
        unsafeAtomicAdd(o + 2, norm * hv.z);
        unsafeAtomicAdd(o + 3, norm * hv.w);
    }
}

// ---------------------------------------------------------------------------
extern "C" void kernel_launch(void* const* d_in, const int* in_sizes, int n_in,
                              void* d_out, int out_size, void* d_ws, size_t ws_size,
                              hipStream_t stream) {
    const float*     x  = (const float*)d_in[0];
    const long long* ei = (const long long*)d_in[1];   // int64 (2, E)
    const float*     W  = (const float*)d_in[2];
    const float*     b  = (const float*)d_in[3];
    float*           out = (float*)d_out;

    const int n = in_sizes[0] / D_FEAT;    // 65536
    const int E = in_sizes[1] / 2;         // 1048576

    // Workspace layout: deg[n] | dinv[n] | h[n*64]
    float* deg  = (float*)d_ws;
    float* dinv = deg + n;
    float* h    = dinv + n;

    const int B = 256;
    gcn_deg_init   <<<(n + B - 1) / B, B, 0, stream>>>(deg, n);
    gcn_deg_count  <<<(E + B - 1) / B, B, 0, stream>>>(ei, deg, E);
    gcn_dinv       <<<(n + B - 1) / B, B, 0, stream>>>(deg, dinv, n);
    gcn_gemm_wmma  <<<n / 16, 128, 0, stream>>>(x, W, h);
    gcn_selfloop_bias<<<(int)(((long long)n * D_FEAT + B - 1) / B), B, 0, stream>>>(h, dinv, b, out, n);
    gcn_scatter    <<<(int)(((long long)E * 16 + B - 1) / B), B, 0, stream>>>(ei, dinv, h, out, E);
}
